// AttentionCritic_85985245266041
// MI455X (gfx1250) — matrix operand order repaired and checked by reading, not compiled, not run on hardware
//
#include <hip/hip_runtime.h>
#include <hip/hip_bf16.h>

// Problem constants (match reference)
#define A_N   15
#define B_N   32768
#define SDIM_ 128
#define ADIM_ 32
#define H_    128
#define NH_   4
#define DH_   32
#define BT    16          // batch tile = WMMA N
#define NWAVE 8           // 256 threads / wave32

#define SMEM_BYTES (76800 + 3 * 61440)   // 261120 B < 320 KB WGP LDS

typedef __attribute__((ext_vector_type(16))) __bf16 v16bf;
typedef __attribute__((ext_vector_type(8)))  __bf16 v8bf;
typedef __attribute__((ext_vector_type(4)))  __bf16 v4bf;
typedef __attribute__((ext_vector_type(8)))  float  v8f;

#define WMMA_BF16(acc, a, b) \
  (acc) = __builtin_amdgcn_wmma_f32_16x16x32_bf16(false, (a), false, (b), (short)0, (acc), false, false)

// ---------------------------------------------------------------------------
// Pre-pass: pack f32 weights into bf16 WMMA A-fragment order.
// One fragment = 16(M) x 32(K); stored as 32 lanes x 16 bf16 (32 B contiguous
// per lane). Per ISA 7.12.2 the lane (l) holds row M = mbase+(l&15) with
// K = k0..k0+7 and k0+16..k0+23, k0 = kbase + 8*(l>=16).
// Main kernel then fetches a fragment with two coalesced global_load_b128.
// heads==0: A[m][k] = src[mat*K*M + k*M + m]          (W stored [K][M])
// heads==1: A[m][k] = src[(m>>5)*K*32 + k*32 + (m&31)] (W stored [NH][K][32])
// ---------------------------------------------------------------------------
__global__ __launch_bounds__(256)
void pack_a_kernel(const float* __restrict__ src, __bf16* __restrict__ dst,
                   int K, int M, int heads, int total) {
  const int gid = blockIdx.x * 256 + threadIdx.x;   // one thread per (frag,lane)
  if (gid >= total) return;
  const int lane = gid & 31;
  const int frag = gid >> 5;
  const int ksteps = K >> 5, mtiles = M >> 4;
  const int ks  = frag % ksteps;
  const int mt  = (frag / ksteps) % mtiles;
  const int mat = frag / (ksteps * mtiles);
  const int m  = mt * 16 + (lane & 15);
  const int k0 = ks * 32 + ((lane >> 4) << 3);
  v8bf lo, hi;
#pragma unroll
  for (int t = 0; t < 8; ++t) {
    const int ka = k0 + t, kb = k0 + 16 + t;
    float va, vb;
    if (heads) {
      const float* base = src + (size_t)(m >> 5) * K * 32 + (m & 31);
      va = base[ka * 32];
      vb = base[kb * 32];
    } else {
      const float* base = src + (size_t)mat * K * M + m;
      va = base[ka * M];
      vb = base[kb * M];
    }
    lo[t] = (__bf16)va;
    hi[t] = (__bf16)vb;
  }
  __bf16* p = dst + ((size_t)gid << 4);
  *(v8bf*)(p)     = lo;
  *(v8bf*)(p + 8) = hi;
}

// Packed A-fragment fetch: 2x global_load_b128, no converts.
__device__ __forceinline__ v16bf load_a_packed(const __bf16* __restrict__ P, int frag) {
  const int lane = threadIdx.x & 31;
  const __bf16* p = P + ((size_t)frag << 9) + (lane << 4);
  v8bf lo = *(const v8bf*)(p);
  v8bf hi = *(const v8bf*)(p + 8);
  return __builtin_shufflevector(lo, hi, 0,1,2,3,4,5,6,7,8,9,10,11,12,13,14,15);
}

// B fragment (32x16 bf16) from LDS activations stored [n][k] (k contiguous).
// Lanes 0-15: col n=lane, K=kbase..kbase+7,kbase+16..kbase+23; lanes 16-31: K=kbase+8..kbase+15,kbase+24..kbase+31.
__device__ __forceinline__ v16bf wmma_b_lds(const __bf16* X, int ldk, int kbase) {
  const int lane = threadIdx.x & 31;
  const int n  = lane & 15;
  const int k0 = kbase + ((lane >> 4) << 3);
  const __bf16* p = X + n * ldk + k0;
  v8bf lo = *(const v8bf*)(p);
  v8bf hi = *(const v8bf*)(p + 16);
  return __builtin_shufflevector(lo, hi, 0,1,2,3,4,5,6,7,8,9,10,11,12,13,14,15);
}

// D tile (16x16 f32): VGPR r -> (M = r + 8*(lane>=16), N = lane&15).
// Each lane owns 8 *contiguous* m's -> bias via 2x float4 loads, output as one
// packed 16B LDS store (4x v_cvt_pk_bf16_f32 + ds_store_b128). Compile-time
// BIAS/ACT: no branches in the epilogue.
template <bool BIAS, bool ACT>
__device__ __forceinline__ void store_d_lds(v8f acc, const float* __restrict__ bias,
                                            __bf16* X, int ldm, int mbase) {
  const int lane = threadIdx.x & 31;
  const int n    = lane & 15;
  const int m0   = mbase + ((lane >> 4) << 3);
  float bb[8];
  if (BIAS) {
    float4 b0 = *(const float4*)&bias[m0];
    float4 b1 = *(const float4*)&bias[m0 + 4];
    bb[0] = b0.x; bb[1] = b0.y; bb[2] = b0.z; bb[3] = b0.w;
    bb[4] = b1.x; bb[5] = b1.y; bb[6] = b1.z; bb[7] = b1.w;
  }
  v8bf o;
#pragma unroll
  for (int r = 0; r < 8; ++r) {
    float v = acc[r];
    if (BIAS) v += bb[r];
    if (ACT) v = (v > 0.f) ? v : 0.01f * v;
    o[r] = (__bf16)v;
  }
  *(v8bf*)&X[n * ldm + m0] = o;   // 16 B, aligned (ldm=128, m0 % 8 == 0)
}

// Packed-weight region sizes (bf16 elements): frag = 512 elems
#define FR 512
#define NF_WENC  (A_N * 8 * 5)   // 600
#define NF_WSENC (A_N * 8 * 4)   // 480
#define NF_HEADS (8 * 4)         // 32 (each of Wk/Wsel/Wv)
#define NF_WC1   (A_N * 8 * 8)   // 960
#define NF_WC2   (A_N * 2 * 4)   // 120

__global__ __launch_bounds__(256, 1)
void attention_critic_kernel(const float* __restrict__ states, const float* __restrict__ actions,
                             const __bf16* __restrict__ Wenc_p,  const float* __restrict__ benc,
                             const __bf16* __restrict__ Wsenc_p, const float* __restrict__ bsenc,
                             const __bf16* __restrict__ Wk_p,    const __bf16* __restrict__ Wsel_p,
                             const __bf16* __restrict__ Wv_p,    const float* __restrict__ bv,
                             const __bf16* __restrict__ Wc1_p,   const float* __restrict__ bc1,
                             const __bf16* __restrict__ Wc2_p,   const float* __restrict__ bc2,
                             float* __restrict__ out) {
  extern __shared__ char smem[];
  // Region plan (bytes):
  //  [0,76800)          sa tile [15][16][160] bf16   -> reused as keys [15][16][128]
  //  [76800,138240)     sa_enc  [15][16][128] bf16   -> reused as sel/ci/h per agent
  //  [138240,199680)    s_enc   [15][16][128] bf16   (persistent)
  //  [199680,261120)    vals    [15][16][128] bf16
  __bf16* sa_lds   = (__bf16*)(smem);
  __bf16* keys_lds = (__bf16*)(smem);
  __bf16* enc_lds  = (__bf16*)(smem + 76800);
  __bf16* senc_lds = (__bf16*)(smem + 76800 + 61440);
  __bf16* vals_lds = (__bf16*)(smem + 76800 + 2 * 61440);
  __bf16* sel_lds  = enc_lds;            // [16][128]  (4 KB)
  __bf16* ci_lds   = enc_lds + 2048;     // [16][256]  (8 KB)
  __bf16* h_lds    = enc_lds + 6144;     // [16][128]  (4 KB)

  const int b0   = blockIdx.x * BT;
  const int wave = threadIdx.x >> 5;
  const v8f vzero = {0.f,0.f,0.f,0.f,0.f,0.f,0.f,0.f};

  // ---- Stage 0: stage (states|actions) -> LDS bf16, layout [a][b][k], ld=160
  // float4 loads (row length 160 and the 128-boundary are both 4-aligned).
  for (int idx = threadIdx.x; idx < (A_N * BT * 160) / 4; idx += 256) {
    const int e = idx * 4;
    const int k = e % 160;
    const int b = (e / 160) % BT;
    const int a = e / (160 * BT);
    float4 v;
    if (k < SDIM_)
      v = *(const float4*)&states[(size_t)a * B_N * SDIM_ + (size_t)(b0 + b) * SDIM_ + k];
    else
      v = *(const float4*)&actions[(size_t)a * B_N * ADIM_ + (size_t)(b0 + b) * ADIM_ + (k - SDIM_)];
    v4bf o;
    o[0] = (__bf16)v.x; o[1] = (__bf16)v.y; o[2] = (__bf16)v.z; o[3] = (__bf16)v.w;
    *(v4bf*)&sa_lds[e] = o;
  }
  __syncthreads();

  // ---- Stage 1: sa_enc = lrelu(sa @ Wenc + benc), s_enc = lrelu(s @ Wsenc + bsenc)
  for (int t = wave; t < A_N * 8; t += NWAVE) {
    const int a  = t >> 3;
    const int mt = t & 7;
    v8f acc = vzero;
#pragma unroll
    for (int ks = 0; ks < 5; ++ks) {   // K = 160
      v16bf af = load_a_packed(Wenc_p, (a * 8 + mt) * 5 + ks);
      v16bf bf = wmma_b_lds(sa_lds + a * BT * 160, 160, ks * 32);
      WMMA_BF16(acc, af, bf);
    }
    store_d_lds<true, true>(acc, benc + a * H_, enc_lds + a * BT * H_, H_, mt << 4);

    v8f acc2 = vzero;
#pragma unroll
    for (int ks = 0; ks < 4; ++ks) {   // K = 128 (states live in k<128 of sa rows)
      v16bf af = load_a_packed(Wsenc_p, (a * 8 + mt) * 4 + ks);
      v16bf bf = wmma_b_lds(sa_lds + a * BT * 160, 160, ks * 32);
      WMMA_BF16(acc2, af, bf);
    }
    store_d_lds<true, true>(acc2, bsenc + a * H_, senc_lds + a * BT * H_, H_, mt << 4);
  }
  __syncthreads();   // sa_lds dead; its region becomes keys_lds

  // ---- Stage 2: keys = sa_enc @ Wk ; vals = lrelu(sa_enc @ Wv + bv)
  for (int t = wave; t < A_N * 8; t += NWAVE) {
    const int a  = t >> 3;
    const int mt = t & 7;
    v8f acck = vzero, accv = vzero;
#pragma unroll
    for (int ks = 0; ks < 4; ++ks) {
      v16bf bf = wmma_b_lds(enc_lds + a * BT * H_, H_, ks * 32);
      v16bf ak = load_a_packed(Wk_p, mt * 4 + ks);
      WMMA_BF16(acck, ak, bf);
      v16bf av = load_a_packed(Wv_p, mt * 4 + ks);
      WMMA_BF16(accv, av, bf);
    }
    store_d_lds<false, false>(acck, nullptr, keys_lds + a * BT * H_, H_, mt << 4);
    store_d_lds<true,  true >(accv, bv,      vals_lds + a * BT * H_, H_, mt << 4);
  }
  __syncthreads();   // enc_lds dead; its region becomes sel/ci/h scratch

  // ---- Stage 3: per-agent attention + critic head
  for (int i = 0; i < A_N; ++i) {
    // (a) sel_i = s_enc[i] @ Wsel   (8 M-tiles, one per wave)
    {
      const int mt = wave;
      v8f acc = vzero;
#pragma unroll
      for (int ks = 0; ks < 4; ++ks) {
        v16bf af = load_a_packed(Wsel_p, mt * 4 + ks);
        v16bf bf = wmma_b_lds(senc_lds + i * BT * H_, H_, ks * 32);
        WMMA_BF16(acc, af, bf);
      }
      store_d_lds<false, false>(acc, nullptr, sel_lds, H_, mt << 4);
    }
    __syncthreads();

    // (b) exclude-self attention, build critic input [s_enc | attn]
    {
      const int b  = threadIdx.x & 15;
      const int n  = (threadIdx.x >> 4) & 3;
      const int dc = threadIdx.x >> 6;            // d-chunk 0..3
      const __bf16* sb = sel_lds + b * H_ + n * DH_;
      float lg[A_N];
      float mx = -3.0e38f;
#pragma unroll
      for (int j = 0; j < A_N; ++j) {
        const __bf16* kb = keys_lds + (j * BT + b) * H_ + n * DH_;
        float s = 0.f;
#pragma unroll
        for (int d = 0; d < DH_; ++d) s += (float)sb[d] * (float)kb[d];
        s *= 0.17677669529663687f;                // 1/sqrt(32)
        if (j == i) s = -1e9f;                    // self mask (matches reference)
        lg[j] = s;
        mx = fmaxf(mx, s);
      }
      float den = 0.f;
#pragma unroll
      for (int j = 0; j < A_N; ++j) { float e = __expf(lg[j] - mx); lg[j] = e; den += e; }
      const float inv = 1.f / den;
#pragma unroll
      for (int d = 0; d < 8; ++d) {
        const int dd = dc * 8 + d;
        float acc = 0.f;
#pragma unroll
        for (int j = 0; j < A_N; ++j)
          acc += lg[j] * (float)vals_lds[(j * BT + b) * H_ + n * DH_ + dd];
        ci_lds[b * (2 * H_) + H_ + n * DH_ + dd] = (__bf16)(acc * inv);
      }
      const int ch = threadIdx.x >> 4;            // 0..15 -> copy s_enc slice (16B)
      *(v8bf*)&ci_lds[b * (2 * H_) + ch * 8] =
          *(const v8bf*)&senc_lds[i * BT * H_ + b * H_ + ch * 8];
    }
    __syncthreads();

    // (c) h = lrelu(ci @ Wc1[i] + bc1[i])   (K = 256)
    {
      const int mt = wave;
      v8f acc = vzero;
#pragma unroll
      for (int ks = 0; ks < 8; ++ks) {
        v16bf af = load_a_packed(Wc1_p, (i * 8 + mt) * 8 + ks);
        v16bf bf = wmma_b_lds(ci_lds, 2 * H_, ks * 32);
        WMMA_BF16(acc, af, bf);
      }
      store_d_lds<true, true>(acc, bc1 + i * H_, h_lds, H_, mt << 4);
    }
    __syncthreads();

    // (d) q = h @ Wc2[i] + bc2[i]  -> d_out   (M = 32: 2 tiles on waves 0,1)
    if (wave < 2) {
      const int mt = wave;
      v8f acc = vzero;
#pragma unroll
      for (int ks = 0; ks < 4; ++ks) {
        v16bf af = load_a_packed(Wc2_p, (i * 2 + mt) * 4 + ks);
        v16bf bf = wmma_b_lds(h_lds, H_, ks * 32);
        WMMA_BF16(acc, af, bf);
      }
      const int lane = threadIdx.x & 31;
      const int n    = lane & 15;
      const int m0   = (mt << 4) + ((lane >> 4) << 3);   // 8 contiguous outputs
      float4 bcl0 = *(const float4*)&bc2[i * ADIM_ + m0];
      float4 bcl1 = *(const float4*)&bc2[i * ADIM_ + m0 + 4];
      float* op = &out[(size_t)i * B_N * ADIM_ + (size_t)(b0 + n) * ADIM_ + m0];
      float4 o0 = {acc[0] + bcl0.x, acc[1] + bcl0.y, acc[2] + bcl0.z, acc[3] + bcl0.w};
      float4 o1 = {acc[4] + bcl1.x, acc[5] + bcl1.y, acc[6] + bcl1.z, acc[7] + bcl1.w};
      *(float4*)(op)     = o0;
      *(float4*)(op + 4) = o1;
    }
    __syncthreads();   // protect sel/ci/h reuse next agent
  }
}

static void launch_pack(const float* src, __bf16* dst, int nmat, int K, int M,
                        int heads, hipStream_t stream) {
  const int total = nmat * (M >> 4) * (K >> 5) * 32;
  pack_a_kernel<<<(total + 255) / 256, 256, 0, stream>>>(src, dst, K, M, heads, total);
}

extern "C" void kernel_launch(void* const* d_in, const int* in_sizes, int n_in,
                              void* d_out, int out_size, void* d_ws, size_t ws_size,
                              hipStream_t stream) {
  const float* states = (const float*)d_in[0];
  const float* actions= (const float*)d_in[1];
  const float* Wenc   = (const float*)d_in[2];
  const float* benc   = (const float*)d_in[3];
  const float* Wsenc  = (const float*)d_in[4];
  const float* bsenc  = (const float*)d_in[5];
  const float* Wk     = (const float*)d_in[6];
  const float* Wsel   = (const float*)d_in[7];
  const float* Wv     = (const float*)d_in[8];
  const float* bv     = (const float*)d_in[9];
  const float* Wc1    = (const float*)d_in[10];
  const float* bc1    = (const float*)d_in[11];
  const float* Wc2    = (const float*)d_in[12];
  const float* bc2    = (const float*)d_in[13];
  float* out = (float*)d_out;

  // Packed-weight workspace layout (bf16), total ~2.2 MB (L2-resident)
  __bf16* w = (__bf16*)d_ws;
  __bf16* Wenc_p  = w;                         // 600 frags
  __bf16* Wsenc_p = Wenc_p  + (size_t)NF_WENC  * FR;
  __bf16* Wk_p    = Wsenc_p + (size_t)NF_WSENC * FR;
  __bf16* Wsel_p  = Wk_p    + (size_t)NF_HEADS * FR;
  __bf16* Wv_p    = Wsel_p  + (size_t)NF_HEADS * FR;
  __bf16* Wc1_p   = Wv_p    + (size_t)NF_HEADS * FR;
  __bf16* Wc2_p   = Wc1_p   + (size_t)NF_WC1   * FR;

  launch_pack(Wenc,  Wenc_p,  A_N, 160,     H_,    0, stream);
  launch_pack(Wsenc, Wsenc_p, A_N, SDIM_,   H_,    0, stream);
  launch_pack(Wk,    Wk_p,    1,   H_,      H_,    1, stream);
  launch_pack(Wsel,  Wsel_p,  1,   H_,      H_,    1, stream);
  launch_pack(Wv,    Wv_p,    1,   H_,      H_,    1, stream);
  launch_pack(Wc1,   Wc1_p,   A_N, 2 * H_,  H_,    0, stream);
  launch_pack(Wc2,   Wc2_p,   A_N, H_,      ADIM_, 0, stream);

  (void)hipFuncSetAttribute((const void*)attention_critic_kernel,
                            hipFuncAttributeMaxDynamicSharedMemorySize, SMEM_BYTES);

  dim3 grid(B_N / BT), block(256);
  attention_critic_kernel<<<grid, block, SMEM_BYTES, stream>>>(
      states, actions, Wenc_p, benc, Wsenc_p, bsenc, Wk_p, Wsel_p, Wv_p, bv,
      Wc1_p, bc1, Wc2_p, bc2, out);
}
